// CompressionAttention_25451976196661
// MI455X (gfx1250) — compile-verified, matrix operand order checked by reading, not run on hardware
//
#include <hip/hip_runtime.h>

// ---------------------------------------------------------------------------
// CompressionAttention for MI455X (gfx1250, wave32, WMMA).
//   Phase 0: f32 -> bf16 (RNE) conversion of x and the four weight matrices.
//   Phase 1: Q/K/V projections via v_wmma_f32_16x16x32_bf16 (f32 accum),
//            double-buffered LDS tiles filled by GLOBAL_LOAD_ASYNC_TO_LDS_B128
//            (ASYNCcnt) so the copy of slab i+1 overlaps the WMMAs on slab i.
//   Phase 2: per-(batch,head) compression scan. 16x2048 logits live in LDS
//            (128 KB -- needs CDNA5's 320KB-per-WGP LDS).
//   Phase 3: output projection via WMMA, bias fused, f32 result to d_out.
// ---------------------------------------------------------------------------

typedef __attribute__((ext_vector_type(16))) __bf16 v16bf;
typedef __attribute__((ext_vector_type(8)))  float  v8f;
typedef __attribute__((ext_vector_type(4)))  int    v4i;

#define SEQ    2048
#define BATCH  2
#define DMODEL 1024
#define NHEADS 16
#define DHEAD  64
#define CLEN   16
#define MROWS  (SEQ * BATCH)   // 4096 rows: row = s*BATCH + b

#define GLOBAL_AS __attribute__((address_space(1)))
#define LDS_AS    __attribute__((address_space(3)))

// ---- f32 -> bf16 round-to-nearest-even -----------------------------------
__device__ __forceinline__ unsigned short f2bf(float f) {
    unsigned int u = __float_as_uint(f);
    unsigned int r = u + 0x7FFFu + ((u >> 16) & 1u);
    return (unsigned short)(r >> 16);
}

__global__ void cvt_f32_bf16(const float* __restrict__ in,
                             unsigned short* __restrict__ out, int n) {
    int i = blockIdx.x * blockDim.x + threadIdx.x;
    if (i < n) out[i] = f2bf(in[i]);
}

// ---- CDNA5 async global->LDS 16B copy (ASYNCcnt) --------------------------
__device__ __forceinline__ void async_copy16(void* lds, const void* gptr) {
#if __has_builtin(__builtin_amdgcn_global_load_async_to_lds_b128)
    __builtin_amdgcn_global_load_async_to_lds_b128(
        (GLOBAL_AS v4i*)gptr, (LDS_AS v4i*)lds, 0, 0);
#else
#warning "global_load_async_to_lds_b128 builtin unavailable; synchronous staging fallback"
    *(uint4*)lds = *(const uint4*)gptr;
#endif
}

__device__ __forceinline__ void wait_async0() {
#if __has_builtin(__builtin_amdgcn_s_wait_asynccnt)
    __builtin_amdgcn_s_wait_asynccnt(0);
#else
#warning "s_wait_asynccnt builtin unavailable; inline asm fallback"
    asm volatile("s_wait_asynccnt 0" ::: "memory");
#endif
}

// ---- WMMA GEMM:  C[M][N] = A[M][K] * B[N][K]^T + bias[N] ------------------
// Both operands row-major with contiguous K => identical 16B-chunk fragment
// loads for A and B (ISA 7.12.2 16-bit A 16x32 layout).
union FragBF { v16bf v; uint4 q[2]; };

__global__ __launch_bounds__(256)
void gemm_abT_bf16(const unsigned short* __restrict__ A,   // [M][K] bf16
                   const unsigned short* __restrict__ Bm,  // [N][K] bf16
                   const float* __restrict__ bias,         // [N]
                   float* __restrict__ C,                  // [M][N] f32
                   int M, int N, int K)
{
    constexpr int BM = 128, BN = 64, BK = 64, PITCH = 80; // 160B pitch, 16B aligned
    __shared__ unsigned short As[2][BM][PITCH];
    __shared__ unsigned short Bs[2][BN][PITCH];

    const int tid   = threadIdx.x;
    const int lane  = tid & 31;
    const int wave  = tid >> 5;          // 8 waves
    const int wm    = wave & 3;          // 4 waves down M (32 rows each)
    const int wn    = wave >> 2;         // 2 waves across N (32 cols each)
    const int l16   = lane & 15;
    const int lhalf = lane >> 4;
    const int m0    = blockIdx.y * BM;
    const int n0    = blockIdx.x * BN;

    v8f acc[2][2] = {};

    auto stage = [&](int buf, int kk) {
        // A tile: 128x64 bf16 = 1024 x 16B chunks, 4 per thread.
        #pragma unroll
        for (int j = 0; j < 4; ++j) {
            int i = tid + j * 256;
            int r = i >> 3, c8 = i & 7;
            async_copy16(&As[buf][r][c8 * 8],
                         A + (size_t)(m0 + r) * K + kk + c8 * 8);
        }
        // B tile: 64x64 bf16 = 512 x 16B chunks, 2 per thread.
        #pragma unroll
        for (int j = 0; j < 2; ++j) {
            int i = tid + j * 256;
            int r = i >> 3, c8 = i & 7;
            async_copy16(&Bs[buf][r][c8 * 8],
                         Bm + (size_t)(n0 + r) * K + kk + c8 * 8);
        }
    };

    stage(0, 0);
    wait_async0();
    __syncthreads();

    const int steps = K / BK;            // 16
    for (int it = 0; it < steps; ++it) {
        const int cur = it & 1;
        if (it + 1 < steps) stage(cur ^ 1, (it + 1) * BK);  // overlap with WMMAs

        #pragma unroll
        for (int kc = 0; kc < 2; ++kc) {                    // two K=32 sub-slabs
            FragBF a[2], b[2];
            #pragma unroll
            for (int mi = 0; mi < 2; ++mi) {
                int r = wm * 32 + mi * 16 + l16;
                a[mi].q[0] = *(const uint4*)&As[cur][r][kc * 32 + lhalf * 8];
                a[mi].q[1] = *(const uint4*)&As[cur][r][kc * 32 + 16 + lhalf * 8];
            }
            #pragma unroll
            for (int ni = 0; ni < 2; ++ni) {
                int r = wn * 32 + ni * 16 + l16;
                b[ni].q[0] = *(const uint4*)&Bs[cur][r][kc * 32 + lhalf * 8];
                b[ni].q[1] = *(const uint4*)&Bs[cur][r][kc * 32 + 16 + lhalf * 8];
            }
            #pragma unroll
            for (int mi = 0; mi < 2; ++mi)
                #pragma unroll
                for (int ni = 0; ni < 2; ++ni)
                    acc[mi][ni] = __builtin_amdgcn_wmma_f32_16x16x32_bf16(
                        false, a[mi].v, false, b[ni].v,
                        (short)0, acc[mi][ni], false, false);
        }

        wait_async0();      // next buffer's async copies done (this wave)
        __syncthreads();    // all waves done reading cur / writing nxt
    }

    // Epilogue: C/D layout -> lanes 0-15: (m=r, n=lane); 16-31: (m=r+8, n=lane-16)
    #pragma unroll
    for (int mi = 0; mi < 2; ++mi)
        #pragma unroll
        for (int ni = 0; ni < 2; ++ni) {
            int n = n0 + wn * 32 + ni * 16 + l16;
            float bb = bias ? bias[n] : 0.0f;
            #pragma unroll
            for (int r = 0; r < 8; ++r) {
                int m = m0 + wm * 32 + mi * 16 + ((lane < 16) ? r : r + 8);
                C[(size_t)m * N + n] = acc[mi][ni][r] + bb;
            }
        }
}

// ---- Compression scan + per-position attention ----------------------------
// One workgroup per group g = b*NHEADS + h. Dynamic LDS ~142 KB.
__global__ __launch_bounds__(256)
void compress_attn(const float* __restrict__ Qf, const float* __restrict__ Kf,
                   const float* __restrict__ Vf, const float* __restrict__ q_c,
                   const float* __restrict__ beta,
                   unsigned short* __restrict__ Obf)
{
    extern __shared__ char smem_raw[];
    float* logits = (float*)smem_raw;          // [16][2048] 128 KB
    float* qc_sh  = logits + CLEN * SEQ;       // [16][64]
    float* num    = qc_sh  + CLEN * DHEAD;     // [16][128] running sum of w*[k||v]
    float* den    = num    + CLEN * 128;       // [16]      running sum of w
    float* ksh    = den    + 16;               // [64]
    float* vsh    = ksh    + 64;               // [64]
    float* qsh    = vsh    + 64;               // [64]
    float* wsh    = qsh    + 64;               // [16]
    float* red    = wsh    + 16;               // [256]
    float* p      = red    + 256;              // [16]
    float* mmax   = p      + 16;               // [16]

    const int g = blockIdx.x;
    const int b = g / NHEADS, h = g % NHEADS;
    const int t = threadIdx.x;
    const int c   = t & 15;    // slot owned in strided work
    const int si0 = t >> 4;

    // qc[h][c][d] = q_c[c][h*64+d]
    for (int i = t; i < CLEN * DHEAD; i += 256) {
        int cc = i >> 6, d = i & 63;
        qc_sh[i] = q_c[cc * DMODEL + h * DHEAD + d];
    }
    __syncthreads();

    // Pass 1: logits[c][s] = qc[c] . k[s], full max over s per slot.
    float lm = -3.0e38f;
    for (int s = si0; s < SEQ; s += 16) {
        const float* krow = Kf + (size_t)(s * BATCH + b) * DMODEL + h * DHEAD;
        float acc = 0.f;
        #pragma unroll
        for (int d = 0; d < DHEAD; ++d) acc += qc_sh[c * DHEAD + d] * krow[d];
        logits[c * SEQ + s] = acc;
        lm = fmaxf(lm, acc);
    }
    red[t] = lm;
    __syncthreads();
    if (t < 16) {
        float mx = -3.0e38f;
        for (int j = 0; j < 16; ++j) mx = fmaxf(mx, red[j * 16 + t]);
        mmax[t] = mx;
    }
    for (int i = t; i < CLEN * 128; i += 256) num[i] = 0.f;
    if (t < 16) den[t] = 0.f;
    __syncthreads();

    const float qscale = 1.0f / (8.0f * __expf(beta[h])); // 1/(sqrt(64)*e^beta)

    // Pass 2: inclusive scan over s, attention against running k_c/v_c.
    for (int s = 0; s < SEQ; ++s) {
        const size_t row = (size_t)(s * BATCH + b);
        if (t < 64)       ksh[t]       = Kf[row * DMODEL + h * DHEAD + t];
        else if (t < 128) vsh[t - 64]  = Vf[row * DMODEL + h * DHEAD + (t - 64)];
        else if (t < 192) qsh[t - 128] = Qf[row * DMODEL + h * DHEAD + (t - 128)] * qscale;
        __syncthreads();

        if (t < 16) {
            float w = __expf(logits[t * SEQ + s] - mmax[t]);
            wsh[t] = w;
            den[t] += w;
        }
        __syncthreads();

        // num[c][0:64] += w[c]*k ; num[c][64:128] += w[c]*v   (2048 fma, 8/thread)
        #pragma unroll
        for (int j = 0; j < 8; ++j) {
            int e  = t + j * 256;
            int cc = e >> 7, d = e & 127;
            float kv = (d < 64) ? ksh[d] : vsh[d - 64];
            num[e] += wsh[cc] * kv;
        }
        __syncthreads();

        // att[c] = (qs . num_k[c]) / den[c] -- 16x16 partial tree
        {
            int part = t >> 4;
            float pa = 0.f;
            #pragma unroll
            for (int j = 0; j < 4; ++j) pa += qsh[part * 4 + j] * num[c * 128 + part * 4 + j];
            red[part * 16 + c] = pa;
        }
        __syncthreads();
        if (t < 16) {
            float a = 0.f;
            for (int j = 0; j < 16; ++j) a += red[j * 16 + t];
            p[t] = a / den[t];
        }
        __syncthreads();
        if (t == 0) {  // 16-way softmax
            float mx = p[0];
            for (int j = 1; j < 16; ++j) mx = fmaxf(mx, p[j]);
            float sum = 0.f;
            for (int j = 0; j < 16; ++j) { float e = __expf(p[j] - mx); p[j] = e; sum += e; }
            float inv = 1.0f / sum;
            for (int j = 0; j < 16; ++j) p[j] *= inv;
        }
        __syncthreads();
        if (t < 64) {  // o[d] = sum_c p[c] * v_c[c][d]
            float o = 0.f;
            #pragma unroll
            for (int cc = 0; cc < 16; ++cc)
                o += p[cc] * (num[cc * 128 + 64 + t] / den[cc]);
            Obf[row * DMODEL + h * DHEAD + t] = f2bf(o);
        }
        __syncthreads();
    }
}

// ---------------------------------------------------------------------------
extern "C" void kernel_launch(void* const* d_in, const int* in_sizes, int n_in,
                              void* d_out, int out_size, void* d_ws, size_t ws_size,
                              hipStream_t stream) {
    (void)in_sizes; (void)n_in; (void)out_size; (void)ws_size;
    const float* x    = (const float*)d_in[0];
    const float* q_c  = (const float*)d_in[1];
    const float* beta = (const float*)d_in[2];
    const float* Wq   = (const float*)d_in[3];
    const float* bq   = (const float*)d_in[4];
    const float* Wk   = (const float*)d_in[5];
    const float* bk   = (const float*)d_in[6];
    const float* Wv   = (const float*)d_in[7];
    const float* bv   = (const float*)d_in[8];
    const float* Wo   = (const float*)d_in[9];
    const float* bo   = (const float*)d_in[10];
    float* out = (float*)d_out;

    char* ws = (char*)d_ws;
    size_t off = 0;
    auto carve = [&](size_t bytes) -> char* {
        char* pp = ws + off;
        off += (bytes + 255) & ~(size_t)255;
        return pp;
    };
    unsigned short* xb  = (unsigned short*)carve((size_t)MROWS  * DMODEL * 2);
    unsigned short* wqb = (unsigned short*)carve((size_t)DMODEL * DMODEL * 2);
    unsigned short* wkb = (unsigned short*)carve((size_t)DMODEL * DMODEL * 2);
    unsigned short* wvb = (unsigned short*)carve((size_t)DMODEL * DMODEL * 2);
    unsigned short* wob = (unsigned short*)carve((size_t)DMODEL * DMODEL * 2);
    float* Qf = (float*)carve((size_t)MROWS * DMODEL * 4);
    float* Kf = (float*)carve((size_t)MROWS * DMODEL * 4);
    float* Vf = (float*)carve((size_t)MROWS * DMODEL * 4);
    unsigned short* Ob = (unsigned short*)carve((size_t)MROWS * DMODEL * 2);

    const int nx = MROWS * DMODEL, nw = DMODEL * DMODEL;
    cvt_f32_bf16<<<(nx + 255) / 256, 256, 0, stream>>>(x,  xb,  nx);
    cvt_f32_bf16<<<(nw + 255) / 256, 256, 0, stream>>>(Wq, wqb, nw);
    cvt_f32_bf16<<<(nw + 255) / 256, 256, 0, stream>>>(Wk, wkb, nw);
    cvt_f32_bf16<<<(nw + 255) / 256, 256, 0, stream>>>(Wv, wvb, nw);
    cvt_f32_bf16<<<(nw + 255) / 256, 256, 0, stream>>>(Wo, wob, nw);

    dim3 ggrid(DMODEL / 64, MROWS / 128);
    gemm_abT_bf16<<<ggrid, 256, 0, stream>>>(xb, wqb, bq, Qf, MROWS, DMODEL, DMODEL);
    gemm_abT_bf16<<<ggrid, 256, 0, stream>>>(xb, wkb, bk, Kf, MROWS, DMODEL, DMODEL);
    gemm_abT_bf16<<<ggrid, 256, 0, stream>>>(xb, wvb, bv, Vf, MROWS, DMODEL, DMODEL);

    const size_t smem =
        (size_t)(CLEN * SEQ + CLEN * DHEAD + CLEN * 128 + 16 + 64 * 3 + 16 + 256 + 16 + 16)
        * sizeof(float);
    (void)hipFuncSetAttribute(reinterpret_cast<const void*>(compress_attn),
                              hipFuncAttributeMaxDynamicSharedMemorySize, (int)smem);
    compress_attn<<<BATCH * NHEADS, 256, smem, stream>>>(Qf, Kf, Vf, q_c, beta, Ob);

    gemm_abT_bf16<<<ggrid, 256, 0, stream>>>(Ob, wob, bo, out, MROWS, DMODEL, DMODEL);
}
